// TransformerBlock_15479062135435
// MI455X (gfx1250) — matrix operand order repaired and checked by reading, not compiled, not run on hardware
//
#include <hip/hip_runtime.h>
#include <hip/hip_bf16.h>
#include <cstddef>
#include <cstdint>

// ---------------------------------------------------------------------------
// Types for CDNA5 WMMA (wave32, 16x16x32 f16 -> f32)
// ---------------------------------------------------------------------------
typedef __attribute__((ext_vector_type(16))) _Float16 v16h;
typedef __attribute__((ext_vector_type(8)))  _Float16 v8h;
typedef __attribute__((ext_vector_type(4)))  _Float16 v4h;
typedef __attribute__((ext_vector_type(8)))  float    v8f;

__device__ __forceinline__ v8f wmma_f16(v16h a, v16h b, v8f c) {
  return __builtin_amdgcn_wmma_f32_16x16x32_f16(
      /*neg_a=*/false, a, /*neg_b=*/false, b,
      /*c_mod=*/(short)0, c, /*reuse_a=*/false, /*reuse_b=*/false);
}

__device__ __forceinline__ v16h frag_from2(v8h lo, v8h hi) {
  v16h r;
#pragma unroll
  for (int i = 0; i < 8; ++i) { r[i] = lo[i]; r[8 + i] = hi[i]; }
  return r;
}

// A-matrix fragment (16 rows x 32 k), tile row-major with leading dim `ld`.
__device__ __forceinline__ v16h load_frag_a(const _Float16* p, int ld) {
  int lane = threadIdx.x & 31;
  const _Float16* r = p + (lane & 15) * ld + ((lane >> 4) << 3);
  return frag_from2(*(const v8h*)r, *(const v8h*)(r + 16));
}

// B-matrix fragment (32 k x 16 cols) from a tile stored TRANSPOSED [col][k].
__device__ __forceinline__ v16h load_frag_bT(const _Float16* p, int ld) {
  int lane = threadIdx.x & 31;
  const _Float16* r = p + (lane & 15) * ld + ((lane >> 4) << 3);
  return frag_from2(*(const v8h*)r, *(const v8h*)(r + 16));
}

// ---------------------------------------------------------------------------
// CDNA5 async global->LDS copies (ASYNCcnt-tracked, no VGPR data pass).
// Per lane, 16 bytes from global vaddr land in LDS at vdst; offset applies to BOTH sides.
// ---------------------------------------------------------------------------
__device__ __forceinline__ unsigned lds_addr32(const void* p) {
  return (unsigned)(uint64_t)p;  // low 32 bits of LDS-aperture address
}
__device__ __forceinline__ void async_b128(unsigned lds, const void* g) {
  uint64_t ga = (uint64_t)g;
  asm volatile("global_load_async_to_lds_b128 %0, %1, off"
               :: "v"(lds), "v"(ga) : "memory");
}
__device__ __forceinline__ void async_b128_off16(unsigned lds, const void* g) {
  uint64_t ga = (uint64_t)g;
  asm volatile("global_load_async_to_lds_b128 %0, %1, off offset:16"
               :: "v"(lds), "v"(ga) : "memory");
}
__device__ __forceinline__ void wait_async0() {
  asm volatile("s_wait_asynccnt 0x0" ::: "memory");
}

// ---------------------------------------------------------------------------
// Problem constants
// ---------------------------------------------------------------------------
#define TB_B 4
#define TB_N 2048
#define TB_C 512
#define TB_H 8
#define TB_D 64
#define TB_F 2048
#define TB_M (TB_B * TB_N)   // 8192 rows

// ---------------------------------------------------------------------------
// LayerNorm (512 cols) -> f16. One wave32 per row, 8 rows / 256-thread block.
// ---------------------------------------------------------------------------
__global__ __launch_bounds__(256) void tb_ln_f16(
    const float* __restrict__ xin, const float* __restrict__ w,
    const float* __restrict__ bb, _Float16* __restrict__ y) {
  int lane = threadIdx.x & 31, wv = threadIdx.x >> 5;
  size_t row = (size_t)blockIdx.x * 8 + wv;
  const float4* xr = (const float4*)(xin + row * TB_C);
  float4 v[4];
  float s = 0.f;
#pragma unroll
  for (int i = 0; i < 4; ++i) {
    v[i] = xr[i * 32 + lane];
    s += v[i].x + v[i].y + v[i].z + v[i].w;
  }
#pragma unroll
  for (int off = 16; off >= 1; off >>= 1) s += __shfl_xor(s, off, 32);
  float mu = s * (1.f / TB_C);
  float ss = 0.f;
#pragma unroll
  for (int i = 0; i < 4; ++i) {
    float a = v[i].x - mu, b2 = v[i].y - mu, c = v[i].z - mu, d = v[i].w - mu;
    ss += a * a + b2 * b2 + c * c + d * d;
  }
#pragma unroll
  for (int off = 16; off >= 1; off >>= 1) ss += __shfl_xor(ss, off, 32);
  float rstd = rsqrtf(ss * (1.f / TB_C) + 1e-5f);
  v4h* yr = (v4h*)(y + row * TB_C);
  const float4* wf = (const float4*)w;
  const float4* bf = (const float4*)bb;
#pragma unroll
  for (int i = 0; i < 4; ++i) {
    float4 w4 = wf[i * 32 + lane], b4 = bf[i * 32 + lane];
    v4h o;
    o[0] = (_Float16)((v[i].x - mu) * rstd * w4.x + b4.x);
    o[1] = (_Float16)((v[i].y - mu) * rstd * w4.y + b4.y);
    o[2] = (_Float16)((v[i].z - mu) * rstd * w4.z + b4.z);
    o[3] = (_Float16)((v[i].w - mu) * rstd * w4.w + b4.w);
    yr[i * 32 + lane] = o;
  }
}

// ---------------------------------------------------------------------------
// Weight convert: W[K][N] f32 -> Wt[N][K] f16 (transposed for B-fragments)
// ---------------------------------------------------------------------------
__global__ __launch_bounds__(256) void tb_convert_wt(
    const float* __restrict__ w, _Float16* __restrict__ wt, int K, int N) {
  int idx = blockIdx.x * 256 + threadIdx.x;
  if (idx < K * N) {
    int n = idx / K, k = idx - n * K;
    wt[idx] = (_Float16)w[(size_t)k * N + n];
  }
}

// ---------------------------------------------------------------------------
// WMMA GEMM: out[M,N] = A[M,K](f16) * W[K,N] (Wt given as [N][K] f16) + bias
// Block tile 128x128, BK=32, double-buffered async global->LDS staging.
// 8 waves (4x2), each computes 32x64 = 2x4 WMMA tiles (8 wmma / k-step).
// EPI 0: f16 store   EPI 1: exact GELU -> f16   EPI 2: resid + gamma*v -> f32
// ---------------------------------------------------------------------------
#define GBM 128
#define GBN 128
#define GBK 32

template <int EPI>
__global__ __launch_bounds__(256) void tb_wmma_gemm(
    const _Float16* __restrict__ A, int lda,
    const _Float16* __restrict__ Bt,          // [N][K]
    const float* __restrict__ bias, int Ncols, int K,
    _Float16* __restrict__ outh, float* __restrict__ outf,
    const float* __restrict__ resid, const float* __restrict__ gamma) {
  __shared__ _Float16 As[2][GBM * GBK];
  __shared__ _Float16 Bs[2][GBN * GBK];
  int t = threadIdx.x;
  int lane = t & 31, w = t >> 5;
  int wm = w & 3, wn = w >> 2;               // 4 x 2 wave grid
  size_t rowBase = (size_t)blockIdx.x * GBM;
  int colBase = blockIdx.y * GBN;

  // per-thread staging coordinates: 16 halves (32B) of A and of Bt each
  int r = t >> 1, cb = (t & 1) << 4;
  const _Float16* gA = A + (rowBase + r) * lda + cb;
  const _Float16* gB = Bt + (size_t)(colBase + r) * K + cb;
  unsigned ldsA0 = lds_addr32(&As[0][r * GBK + cb]);
  unsigned ldsB0 = lds_addr32(&Bs[0][r * GBK + cb]);

  auto stage = [&](int buf, int k0) {
    unsigned bo = (unsigned)buf * (GBM * GBK * 2);  // byte offset of buffer 1
    async_b128(ldsA0 + bo, gA + k0);
    async_b128_off16(ldsA0 + bo, gA + k0);
    async_b128(ldsB0 + bo, gB + k0);
    async_b128_off16(ldsB0 + bo, gB + k0);
  };

  v8f acc[2][4] = {};
  const int nt = K / GBK;
  stage(0, 0);  // prologue

#pragma unroll 2
  for (int it = 0; it < nt; ++it) {
    wait_async0();       // tile `it` resident (own asyncs); barrier makes all visible
    __syncthreads();
    if (it + 1 < nt) stage((it + 1) & 1, (it + 1) * GBK);  // overlap with compute

    const _Float16* as = &As[it & 1][0];
    const _Float16* bs = &Bs[it & 1][0];
    v16h a0 = load_frag_a(as + (wm * 32) * GBK, GBK);
    v16h a1 = load_frag_a(as + (wm * 32 + 16) * GBK, GBK);
#pragma unroll
    for (int ntile = 0; ntile < 4; ++ntile) {
      v16h bfr = load_frag_bT(bs + (wn * 64 + ntile * 16) * GBK, GBK);
      acc[0][ntile] = wmma_f16(a0, bfr, acc[0][ntile]);
      acc[1][ntile] = wmma_f16(a1, bfr, acc[1][ntile]);
    }
  }

  // Epilogue. C/D layout: lane holds col = lane&15; vgpr i holds row i+hi8.
  int ci = lane & 15, hi8 = (lane >> 4) << 3;
#pragma unroll
  for (int mt = 0; mt < 2; ++mt)
#pragma unroll
    for (int ntile = 0; ntile < 4; ++ntile) {
      int col = colBase + wn * 64 + ntile * 16 + ci;
      float bv = bias[col];
#pragma unroll
      for (int i = 0; i < 8; ++i) {
        size_t row = rowBase + wm * 32 + mt * 16 + i + hi8;
        float v = acc[mt][ntile][i] + bv;
        size_t off = row * (size_t)Ncols + col;
        if (EPI == 0) {
          outh[off] = (_Float16)v;
        } else if (EPI == 1) {
          float g = 0.5f * v * (1.0f + erff(v * 0.70710678118f));
          outh[off] = (_Float16)g;
        } else {
          outf[off] = resid[off] + gamma[col] * v;
        }
      }
    }
}

// ---------------------------------------------------------------------------
// Flash attention: per (b,h) and 64-query tile. Online softmax, WMMA for
// S = Q K^T and O += P V. qkv layout: [B*N][1536] with q|k|v at +0/+512/+1024.
// K tiles: double-buffered async global->LDS. V tiles: double-buffered via
// VGPRs (next tile's global loads issued under current tile's compute), then
// transposed into LDS at iteration top.
// ---------------------------------------------------------------------------
__global__ __launch_bounds__(256) void tb_attn(
    const _Float16* __restrict__ qkv, const float* __restrict__ bias,
    _Float16* __restrict__ outp) {
  __shared__ _Float16 Ks[2][64 * 64];         // [key][d]
  __shared__ _Float16 Vt[2][64 * 64];         // [d][key] (transposed)
  __shared__ _Float16 Pw[8 * 16 * 64];        // per-wave P scratch [16][64]
  int t = threadIdx.x, lane = t & 31, w = t >> 5;
  int bh = blockIdx.y, b = bh >> 3, h = bh & 7;
  int q0 = blockIdx.x * 64;
  int ci = lane & 15, hi8 = (lane >> 4) << 3;

  // Q fragments (16 rows x 64 d -> two A-frags), loaded once from global.
  size_t qrow_l = (size_t)(b * TB_N + q0 + w * 16 + ci);
  const _Float16* qp = qkv + qrow_l * (3 * TB_C) + h * TB_D;
  int khiA = (lane >> 4) << 3;
  v16h qa0 = frag_from2(*(const v8h*)(qp + khiA), *(const v8h*)(qp + khiA + 16));
  v16h qa1 = frag_from2(*(const v8h*)(qp + 32 + khiA), *(const v8h*)(qp + 32 + khiA + 16));

  // staging coordinates (16 halves / thread for K and for V)
  int rk = t >> 2, cbk = (t & 3) << 4;
  const _Float16* gK =
      qkv + (size_t)(b * TB_N + rk) * (3 * TB_C) + TB_C + h * TB_D + cbk;
  const _Float16* gV =
      qkv + (size_t)(b * TB_N + rk) * (3 * TB_C) + 2 * TB_C + h * TB_D + cbk;
  unsigned ldsK0 = lds_addr32(&Ks[0][rk * 64 + cbk]);
  const size_t jstride = (size_t)64 * (3 * TB_C);   // 64 qkv rows

  v8f O[4] = {};
  float m8[8], l8[8];
#pragma unroll
  for (int i = 0; i < 8; ++i) { m8[i] = -1e30f; l8[i] = 0.f; }

  // prologue: async-stage K tile 0; pull V tile 0 into VGPRs
  async_b128(ldsK0, gK);
  async_b128_off16(ldsK0, gK);
  v8h vlo = *(const v8h*)gV, vhi = *(const v8h*)(gV + 8);

#pragma unroll 2
  for (int it = 0; it < TB_N / 64; ++it) {
    int j = it * 64;
    int cur = it & 1;
    wait_async0();                           // K[cur] resident (own part)
    // commit V VGPRs -> Vt[cur] (transposed)
    {
      _Float16* vt = &Vt[cur][0];
#pragma unroll
      for (int d = 0; d < 8; ++d) {
        vt[(cbk + d) * 64 + rk] = vlo[d];
        vt[(cbk + 8 + d) * 64 + rk] = vhi[d];
      }
    }
    __syncthreads();                         // K[cur], Vt[cur] published

    if (it + 1 < TB_N / 64) {                // pipeline next tile under compute
      size_t jn = (size_t)(j + 64) * (3 * TB_C);
      unsigned bo = (unsigned)((it + 1) & 1) * (64 * 64 * 2);
      async_b128(ldsK0 + bo, gK + jn);
      async_b128_off16(ldsK0 + bo, gK + jn);
      vlo = *(const v8h*)(gV + jn);
      vhi = *(const v8h*)(gV + jn + 8);
      __builtin_prefetch(gK + jn + jstride, 0, 1);   // 2 tiles ahead into cache
      __builtin_prefetch(gV + jn + jstride, 0, 1);
    }

    // S = Q K^T over d (two 32-wide k-steps), 4 key subtiles of 16.
    const _Float16* ks = &Ks[cur][0];
    v8f s[4];
#pragma unroll
    for (int kt = 0; kt < 4; ++kt) {
      v16h bk0 = load_frag_bT(ks + (kt * 16) * 64, 64);        // d 0..31
      v16h bk1 = load_frag_bT(ks + (kt * 16) * 64 + 32, 64);   // d 32..63
      v8f z = {};
      z = wmma_f16(qa0, bk0, z);
      s[kt] = wmma_f16(qa1, bk1, z);
    }

    // scale + rel-pos bias
#pragma unroll
    for (int kt = 0; kt < 4; ++kt) {
      int kc = j + kt * 16 + ci;
#pragma unroll
      for (int i = 0; i < 8; ++i) {
        int qr = q0 + w * 16 + i + hi8;
        s[kt][i] = s[kt][i] * 0.125f +
                   bias[((size_t)h * TB_N + qr) * TB_N + kc];
      }
    }

    // Online softmax: per-row stats live in the 16-lane half-group.
#pragma unroll
    for (int i = 0; i < 8; ++i) {
      float mx = fmaxf(fmaxf(s[0][i], s[1][i]), fmaxf(s[2][i], s[3][i]));
#pragma unroll
      for (int off = 8; off >= 1; off >>= 1) mx = fmaxf(mx, __shfl_xor(mx, off, 32));
      float mn = fmaxf(m8[i], mx);
      float alpha = __expf(m8[i] - mn);
      m8[i] = mn;
      float rs = 0.f;
#pragma unroll
      for (int kt = 0; kt < 4; ++kt) {
        float p = __expf(s[kt][i] - mn);
        s[kt][i] = p;
        rs += p;
      }
#pragma unroll
      for (int off = 8; off >= 1; off >>= 1) rs += __shfl_xor(rs, off, 32);
      l8[i] = l8[i] * alpha + rs;
#pragma unroll
      for (int dt = 0; dt < 4; ++dt) O[dt][i] *= alpha;
    }

    // Transpose P (C-layout) -> A-layout through wave-private LDS.
    _Float16* pw = Pw + w * 16 * 64;
#pragma unroll
    for (int kt = 0; kt < 4; ++kt)
#pragma unroll
      for (int i = 0; i < 8; ++i)
        pw[(i + hi8) * 64 + kt * 16 + ci] = (_Float16)s[kt][i];

    v16h pa0 = load_frag_a(pw, 64);        // keys 0..31
    v16h pa1 = load_frag_a(pw + 32, 64);   // keys 32..63
    const _Float16* vt = &Vt[cur][0];
#pragma unroll
    for (int dt = 0; dt < 4; ++dt) {
      v16h bv0 = load_frag_bT(vt + (dt * 16) * 64, 64);       // keys 0..31
      v16h bv1 = load_frag_bT(vt + (dt * 16) * 64 + 32, 64);  // keys 32..63
      O[dt] = wmma_f16(pa0, bv0, O[dt]);
      O[dt] = wmma_f16(pa1, bv1, O[dt]);
    }
  }

  // Normalize and store to attn buffer [M][512] at head offset.
#pragma unroll
  for (int dt = 0; dt < 4; ++dt)
#pragma unroll
    for (int i = 0; i < 8; ++i) {
      size_t qr = (size_t)(b * TB_N + q0 + w * 16 + i + hi8);
      outp[qr * TB_C + h * TB_D + dt * 16 + ci] = (_Float16)(O[dt][i] / l8[i]);
    }
}

// ---------------------------------------------------------------------------
// Launch: workspace layout (bytes), with aliasing of dead buffers.
// ---------------------------------------------------------------------------
extern "C" void kernel_launch(void* const* d_in, const int* in_sizes, int n_in,
                              void* d_out, int out_size, void* d_ws,
                              size_t ws_size, hipStream_t stream) {
  (void)in_sizes; (void)n_in; (void)out_size; (void)ws_size;
  const float* x        = (const float*)d_in[0];
  const float* rel_bias = (const float*)d_in[1];
  const float* qkv_w    = (const float*)d_in[2];
  const float* qkv_b    = (const float*)d_in[3];
  const float* out_w    = (const float*)d_in[4];
  const float* out_b    = (const float*)d_in[5];
  const float* ln1_w    = (const float*)d_in[6];
  const float* ln1_b    = (const float*)d_in[7];
  const float* ln2_w    = (const float*)d_in[8];
  const float* ln2_b    = (const float*)d_in[9];
  const float* ffn_w1   = (const float*)d_in[10];
  const float* ffn_b1   = (const float*)d_in[11];
  const float* ffn_w2   = (const float*)d_in[12];
  const float* ffn_b2   = (const float*)d_in[13];
  const float* gamma1   = (const float*)d_in[14];
  const float* gamma2   = (const float*)d_in[15];

  char* ws = (char*)d_ws;
  _Float16* wqkv_t = (_Float16*)(ws + 0x0000000);   // 1.5 MB
  _Float16* wout_t = (_Float16*)(ws + 0x0180000);   // 0.5 MB
  _Float16* w1_t   = (_Float16*)(ws + 0x0200000);   // 2 MB
  _Float16* w2_t   = (_Float16*)(ws + 0x0400000);   // 2 MB
  float*    x1     = (float*)   (ws + 0x0600000);   // 16 MB
  _Float16* y_h    = (_Float16*)(ws + 0x1600000);   // 8 MB (LN1 & LN2)
  _Float16* qkv_h  = (_Float16*)(ws + 0x1E00000);   // 24 MB
  _Float16* attn_h = (_Float16*)(ws + 0x3600000);   // 8 MB
  _Float16* h_h    = (_Float16*)(ws + 0x1E00000);   // 32 MB, aliases qkv+attn (dead)

  // 1) weights fp32 -> f16 transposed [N][K]
  tb_convert_wt<<<(TB_C * 3 * TB_C + 255) / 256, 256, 0, stream>>>(qkv_w, wqkv_t, TB_C, 3 * TB_C);
  tb_convert_wt<<<(TB_C * TB_C + 255) / 256, 256, 0, stream>>>(out_w, wout_t, TB_C, TB_C);
  tb_convert_wt<<<(TB_C * TB_F + 255) / 256, 256, 0, stream>>>(ffn_w1, w1_t, TB_C, TB_F);
  tb_convert_wt<<<(TB_F * TB_C + 255) / 256, 256, 0, stream>>>(ffn_w2, w2_t, TB_F, TB_C);

  // 2) y = LN1(x) -> f16
  tb_ln_f16<<<TB_M / 8, 256, 0, stream>>>(x, ln1_w, ln1_b, y_h);

  // 3) qkv = y @ qkv_w + b  -> f16 [M,1536]
  tb_wmma_gemm<0><<<dim3(TB_M / GBM, (3 * TB_C) / GBN), 256, 0, stream>>>(
      y_h, TB_C, wqkv_t, qkv_b, 3 * TB_C, TB_C, qkv_h, nullptr, nullptr, nullptr);

  // 4) flash attention with rel-pos bias -> f16 [M,512]
  tb_attn<<<dim3(TB_N / 64, TB_B * TB_H), 256, 0, stream>>>(qkv_h, rel_bias, attn_h);

  // 5) x1 = x + gamma1 * (attn @ out_w + out_b)  -> f32
  tb_wmma_gemm<2><<<dim3(TB_M / GBM, TB_C / GBN), 256, 0, stream>>>(
      attn_h, TB_C, wout_t, out_b, TB_C, TB_C, nullptr, x1, x, gamma1);

  // 6) y2 = LN2(x1) -> f16 (reuses y_h)
  tb_ln_f16<<<TB_M / 8, 256, 0, stream>>>(x1, ln2_w, ln2_b, y_h);

  // 7) h = gelu(y2 @ w1 + b1) -> f16 [M,2048]
  tb_wmma_gemm<1><<<dim3(TB_M / GBM, TB_F / GBN), 256, 0, stream>>>(
      y_h, TB_C, w1_t, ffn_b1, TB_F, TB_C, h_h, nullptr, nullptr, nullptr);

  // 8) out = x1 + gamma2 * (h @ w2 + b2) -> f32 d_out
  tb_wmma_gemm<2><<<dim3(TB_M / GBM, TB_C / GBN), 256, 0, stream>>>(
      h_h, TB_F, w2_t, ffn_b2, TB_C, TB_F, nullptr, (float*)d_out, x1, gamma2);
}
